// PositionConv_21912923144643
// MI455X (gfx1250) — matrix operand order repaired and checked by reading, not compiled, not run on hardware
//
#include <hip/hip_runtime.h>
#include <hip/hip_bf16.h>
#include <stdint.h>

typedef __attribute__((ext_vector_type(16))) __bf16 v16bf;
typedef __attribute__((ext_vector_type(8)))  __bf16 v8bf;
typedef __attribute__((ext_vector_type(8)))  float  v8f;
typedef int v4i __attribute__((vector_size(16)));

#define BATCH 4
#define CIN   64
#define COUT  64
#define RDIM  (CIN * 27)             // 1728
#define HW    4096
#define DHW   262144

// LDS patch: [6 h-rows][66 padded w][ci padded to 72]
#define PROWS 6
#define PW    66
#define CPAD  72                     // 144B stride: 16B aligned, conflict-free
#define PATCH_ELEMS (PROWS * PW * CPAD)

// ---------------------------------------------------------------------------
// async / wait helpers (builtin if present, inline CDNA5 asm otherwise)
// ---------------------------------------------------------------------------
#if defined(__has_builtin) && __has_builtin(__builtin_amdgcn_s_wait_asynccnt)
#define WAIT_ASYNC(n) __builtin_amdgcn_s_wait_asynccnt(n)
#else
#define WAIT_ASYNC(n) asm volatile("s_wait_asynccnt %0" ::"i"(n) : "memory")
#endif

static __device__ __forceinline__ void async_ld_b128(const __bf16* g, __bf16* l) {
#if defined(__has_builtin) && __has_builtin(__builtin_amdgcn_global_load_async_to_lds_b128)
    __builtin_amdgcn_global_load_async_to_lds_b128(
        (__attribute__((address_space(1))) v4i*)g,
        (__attribute__((address_space(3))) v4i*)l, 0, 0);
#else
    unsigned loff = (unsigned)(size_t)(__attribute__((address_space(3))) char*)l;
    unsigned long long ga = (unsigned long long)(uintptr_t)g;
    asm volatile("global_load_async_to_lds_b128 %0, %1, off"
                 ::"v"(loff), "v"(ga) : "memory");
#endif
}

// ---------------------------------------------------------------------------
// Kernel 0: x [b][ci][d][h][w] f32  ->  xT [b][d][h][w][ci] bf16 (LDS-tiled
// transpose so both global sides are coalesced). Block 0 also zeros the
// 8KB zero-row block used for padded async loads.
// ---------------------------------------------------------------------------
__global__ __launch_bounds__(256)
void cvt_transpose_kernel(const float* __restrict__ x,
                          __bf16* __restrict__ xT,
                          __bf16* __restrict__ zblk) {
    __shared__ __bf16 tile[64 * CPAD];          // [w][ci pad 72]
    int blk = blockIdx.x;                        // (b*64 + d)*64 + h
    int t   = threadIdx.x;

    const float* src = x + (size_t)(blk >> 12) * CIN * DHW    // batch
                         + (size_t)((blk >> 6) & 63) * HW     // d
                         + (size_t)(blk & 63) * 64;           // h
    int w  = t & 63;
    int c0 = t >> 6;                             // 0..3
#pragma unroll
    for (int r = 0; r < 16; ++r) {
        int ci = c0 * 16 + r;
        tile[w * CPAD + ci] = (__bf16)src[(size_t)ci * DHW + w];
    }
    __syncthreads();

    __bf16* dst = xT + (size_t)blk * (64 * CIN);
    int wo = t >> 2, cb = (t & 3) * 16;
    *(v8bf*)(dst + wo * 64 + cb)     = *(const v8bf*)&tile[wo * CPAD + cb];
    *(v8bf*)(dst + wo * 64 + cb + 8) = *(const v8bf*)&tile[wo * CPAD + cb + 8];

    if (blk == 0) {
        v8bf z;
#pragma unroll
        for (int i = 0; i < 8; ++i) z[i] = (__bf16)0.0f;
        for (int e = t; e < 4096 / 8; e += 256) *(v8bf*)(zblk + e * 8) = z;
    }
}

// ---------------------------------------------------------------------------
// Kernel 1: dynamic weights -> AW[b][co][r'], r' = ((kd*3+kh)*3+kw)*64 + ci
// ---------------------------------------------------------------------------
__global__ __launch_bounds__(256)
void dyn_weight_kernel(const float* __restrict__ pos_emb,
                       const float* __restrict__ W_pos,
                       const float* __restrict__ b_pos,
                       __bf16* __restrict__ AW) {
    int idx = blockIdx.x * 256 + threadIdx.x;    // B*COUT*RDIM = 442368
    if (idx >= BATCH * COUT * RDIM) return;
    int b   = idx / (COUT * RDIM);
    int rem = idx % (COUT * RDIM);
    int co  = rem / RDIM;
    int rp  = rem % RDIM;
    int ci  = rp & 63;
    int tap = rp >> 6;
    int kw  = tap % 3, kh = (tap / 3) % 3, kd = tap / 9;
    int flat = co * RDIM + ci * 27 + kd * 9 + kh * 3 + kw;
    int n = flat / 576, f = flat % 576;
    const float* pe = pos_emb + ((size_t)b * 192 + n) * 64;
    float acc = b_pos[f];
#pragma unroll 8
    for (int c = 0; c < 64; ++c) acc += pe[c] * W_pos[c * 576 + f];
    AW[idx] = (__bf16)acc;
}

// ---------------------------------------------------------------------------
// Shared compute: 18 K-steps (kh,kw,ci-half) of WMMA over one kd slice.
// ---------------------------------------------------------------------------
static __device__ __forceinline__ void conv_compute_kd(
    const __bf16* __restrict__ buf, const __bf16* __restrict__ arow,
    int kd, int g, int g2, int nlo, v8f acc[8]) {
#pragma unroll
    for (int kh = 0; kh < 3; ++kh) {
#pragma unroll
        for (int kw = 0; kw < 3; ++kw) {
#pragma unroll
            for (int cc = 0; cc < 2; ++cc) {
                int r0 = ((kd * 3 + kh) * 3 + kw) * 64 + cc * 32;
                // A frag: lanes 0-15 K {0..7,16..23}; lanes 16-31 K {8..15,24..31}
                v8bf a_lo = *(const v8bf*)(arow + r0 + g * 8);
                v8bf a_hi = *(const v8bf*)(arow + r0 + 16 + g * 8);
                v16bf afrag;
#pragma unroll
                for (int i = 0; i < 8; ++i) { afrag[i] = a_lo[i]; afrag[i + 8] = a_hi[i]; }
#pragma unroll
                for (int t = 0; t < 8; ++t) {
                    int n  = g2 * 8 + t;
                    int ho = n >> 2;
                    int w0 = (n & 3) << 4;
                    // B frag: lane half g holds K = ci cc*32 + {g*8..g*8+7, 16+g*8..16+g*8+7}
                    const __bf16* bp =
                        &buf[((ho + kh) * PW + w0 + nlo + kw) * CPAD + cc * 32 + g * 8];
                    v8bf b_lo = *(const v8bf*)bp;
                    v8bf b_hi = *(const v8bf*)(bp + 16);
                    v16bf bfrag;
#pragma unroll
                    for (int i = 0; i < 8; ++i) { bfrag[i] = b_lo[i]; bfrag[i + 8] = b_hi[i]; }
                    acc[t] = __builtin_amdgcn_wmma_f32_16x16x32_bf16(
                        false, afrag, false, bfrag, (short)0, acc[t], false, false);
                }
            }
        }
    }
}

static __device__ __forceinline__ void conv_store(
    float* __restrict__ out, const v8f acc[8],
    int b, int d, int hbase, int mtile, int g, int g2, int nlo) {
#pragma unroll
    for (int t = 0; t < 8; ++t) {
        int n  = g2 * 8 + t;
        int ho = n >> 2;
        int w0 = (n & 3) << 4;
        size_t base = ((size_t)b * COUT + mtile * 16 + g * 8) * DHW
                    + (size_t)d * HW + (size_t)(hbase + ho) * 64 + (w0 + nlo);
#pragma unroll
        for (int v = 0; v < 8; ++v) out[base + (size_t)v * DHW] = acc[t][v];
    }
}

// ---------------------------------------------------------------------------
// Kernel 2a: async double-buffered conv (xT bf16 staged by
// GLOBAL_LOAD_ASYNC_TO_LDS_B128; kd+1 slice prefetched during WMMA on kd).
// Every wave issues exactly 12 asyncs per fill (OOB rows pull from a zero
// block) so s_wait_asynccnt 12 is a valid pipeline wait.
// ---------------------------------------------------------------------------
static __device__ __forceinline__ void issue_fill(
    __bf16* __restrict__ buf, const __bf16* __restrict__ xTb,
    const __bf16* __restrict__ zblk, int zd, int hbase, int tid) {
    bool dok = (unsigned)zd < 64u;
#pragma unroll
    for (int i = 0; i < 12; ++i) {               // 3072 b128 chunks / 256 thr
        int c   = i * 256 + tid;
        int ci8 = (c & 7) * 8;
        int w   = (c >> 3) & 63;
        int r   = c >> 9;                        // 0..5
        int zh  = hbase + r - 1;
        const __bf16* gsrc =
            (dok && (unsigned)zh < 64u)
                ? xTb + (((size_t)zd * 64 + zh) * 64 + w) * 64 + ci8
                : zblk + w * 64 + ci8;
        async_ld_b128(gsrc, &buf[(r * PW + w + 1) * CPAD + ci8]);
    }
}

__global__ __launch_bounds__(256)
void conv_wmma_async_kernel(const __bf16* __restrict__ xT,
                            const __bf16* __restrict__ zblk,
                            const __bf16* __restrict__ AW,
                            float* __restrict__ out) {
    __shared__ __bf16 patch[2 * PATCH_ELEMS];    // double buffer: 2 x 57KB

    int blk   = blockIdx.x;
    int htile = blk & 15;
    int d     = (blk >> 4) & 63;
    int b     = blk >> 10;
    int hbase = htile * 4;

    int tid   = threadIdx.x;
    int lane  = tid & 31;
    int mtile = (tid >> 5) & 3;
    int g2    = tid >> 7;
    int g     = lane >> 4;
    int nlo   = lane & 15;

    const __bf16* xTb  = xT + (size_t)b * DHW * CIN;
    const __bf16* arow = AW + ((size_t)(b * COUT + mtile * 16 + nlo)) * RDIM;
    __bf16* buf0 = patch;
    __bf16* buf1 = patch + PATCH_ELEMS;

    // zero the w-pad columns (wp==0 and wp==65) of both buffers, once
    {
        v8bf z;
#pragma unroll
        for (int i = 0; i < 8; ++i) z[i] = (__bf16)0.0f;
        for (int e = tid; e < PROWS * 2 * 8; e += 256) {
            int ci8 = (e & 7) * 8;
            int col = ((e >> 3) & 1) ? (PW - 1) : 0;
            int r   = e >> 4;
            *(v8bf*)&buf0[(r * PW + col) * CPAD + ci8] = z;
            *(v8bf*)&buf1[(r * PW + col) * CPAD + ci8] = z;
        }
    }

    v8f acc[8];
#pragma unroll
    for (int t = 0; t < 8; ++t)
#pragma unroll
        for (int v = 0; v < 8; ++v) acc[t][v] = 0.0f;

    issue_fill(buf0, xTb, zblk, d - 1, hbase, tid);      // kd=0
    issue_fill(buf1, xTb, zblk, d,     hbase, tid);      // kd=1
    WAIT_ASYNC(12);                                      // kd=0 patch landed
    __syncthreads();
    conv_compute_kd(buf0, arow, 0, g, g2, nlo, acc);
    __syncthreads();                                     // buf0 free
    issue_fill(buf0, xTb, zblk, d + 1, hbase, tid);      // kd=2
    WAIT_ASYNC(12);                                      // kd=1 patch landed
    __syncthreads();
    conv_compute_kd(buf1, arow, 1, g, g2, nlo, acc);
    WAIT_ASYNC(0);                                       // kd=2 patch landed
    __syncthreads();
    conv_compute_kd(buf0, arow, 2, g, g2, nlo, acc);

    conv_store(out, acc, b, d, hbase, mtile, g, g2, nlo);
}

// ---------------------------------------------------------------------------
// Kernel 2b: direct-fill fallback (round-1 path) if ws is too small for xT.
// ---------------------------------------------------------------------------
__global__ __launch_bounds__(256)
void conv_wmma_direct_kernel(const float* __restrict__ x,
                             const __bf16* __restrict__ AW,
                             float* __restrict__ out) {
    __shared__ __bf16 patch[PATCH_ELEMS];

    int blk   = blockIdx.x;
    int htile = blk & 15;
    int d     = (blk >> 4) & 63;
    int b     = blk >> 10;
    int hbase = htile * 4;

    int tid   = threadIdx.x;
    int lane  = tid & 31;
    int mtile = (tid >> 5) & 3;
    int g2    = tid >> 7;
    int g     = lane >> 4;
    int nlo   = lane & 15;

    const float* xb = x + (size_t)b * CIN * DHW;
    const __bf16* arow = AW + ((size_t)(b * COUT + mtile * 16 + nlo)) * RDIM;

    v8f acc[8];
#pragma unroll
    for (int t = 0; t < 8; ++t)
#pragma unroll
        for (int v = 0; v < 8; ++v) acc[t][v] = 0.0f;

    for (int kd = 0; kd < 3; ++kd) {
        if (kd) __syncthreads();
        int zd = d + kd - 1;
        bool dok = (unsigned)zd < 64u;
        for (int e = tid; e < PROWS * PW * CIN; e += 256) {
            int wp  = e % PW;
            int r   = e / PW;
            int ci  = r & 63;
            int row = r >> 6;
            int iw  = wp - 1;
            int zh  = hbase + row - 1;
            float v = 0.0f;
            if (dok && (unsigned)iw < 64u && (unsigned)zh < 64u)
                v = xb[((size_t)ci * 64 + zd) * HW + zh * 64 + iw];
            patch[(row * PW + wp) * CPAD + ci] = (__bf16)v;
        }
        __syncthreads();
        conv_compute_kd(patch, arow, kd, g, g2, nlo, acc);
    }
    conv_store(out, acc, b, d, hbase, mtile, g, g2, nlo);
}

// ---------------------------------------------------------------------------
extern "C" void kernel_launch(void* const* d_in, const int* in_sizes, int n_in,
                              void* d_out, int out_size, void* d_ws, size_t ws_size,
                              hipStream_t stream) {
    const float* x       = (const float*)d_in[0];
    const float* pos_emb = (const float*)d_in[1];
    const float* W_pos   = (const float*)d_in[2];
    const float* b_pos   = (const float*)d_in[3];
    float*       out     = (float*)d_out;

    const size_t XT_OFF   = 1u << 20;                         // AW in [0, 884736)
    const size_t XT_BYTES = (size_t)BATCH * DHW * CIN * 2;    // 128 MB
    const size_t Z_OFF    = XT_OFF + XT_BYTES;
    const size_t NEED     = Z_OFF + 8192;

    __bf16* AW = (__bf16*)d_ws;
    int wthreads = BATCH * COUT * RDIM;
    dyn_weight_kernel<<<(wthreads + 255) / 256, 256, 0, stream>>>(
        pos_emb, W_pos, b_pos, AW);

    int blocks = BATCH * 64 * 16;                             // 4096
    if (ws_size >= NEED) {
        __bf16* xT   = (__bf16*)((char*)d_ws + XT_OFF);
        __bf16* zblk = (__bf16*)((char*)d_ws + Z_OFF);
        cvt_transpose_kernel<<<BATCH * 64 * 64, 256, 0, stream>>>(x, xT, zblk);
        conv_wmma_async_kernel<<<blocks, 256, 0, stream>>>(xT, zblk, AW, out);
    } else {
        conv_wmma_direct_kernel<<<blocks, 256, 0, stream>>>(x, AW, out);
    }
}
